// TacticalGNN_19078244729008
// MI455X (gfx1250) — compile-verified, matrix operand order checked by reading, not run on hardware
//
#include <hip/hip_runtime.h>
#include <hip/hip_bf16.h>
#include <math.h>

// ---------------------------------------------------------------------------
// TacticalGNN forward for MI455X (gfx1250, wave32, WMMA bf16).
// GEMMs run through one WMMA kernel with pre-transposed bf16 weights:
//   C[M,256] = act( concat_K(A1[gather],A2) @ W[K,256] + bias )
// MT = M-tiles per block (B-fragment register reuse); LDS double-buffered;
// all 4 B fragments batch-loaded per K-step so the WMMA chain isn't
// serialized behind per-tile loads.
// ---------------------------------------------------------------------------

typedef __attribute__((ext_vector_type(16))) __bf16 v16bf;
typedef __attribute__((ext_vector_type(8)))  __bf16 v8bf;
typedef __attribute__((ext_vector_type(4)))  __bf16 v4bf;
typedef __attribute__((ext_vector_type(8)))  float  v8f;

#define HID 256
#define NNODE 512
#define NEDGE 16384

// ---------------------------------------------------------------------------
// fp32 [K,256] row-major  ->  bf16 [256,K] row-major (transpose + convert)
// ---------------------------------------------------------------------------
__global__ void convert_wt_kernel(const float* __restrict__ W, __bf16* __restrict__ Wt, int K)
{
    int idx = blockIdx.x * blockDim.x + threadIdx.x;
    if (idx >= K * HID) return;
    int k = idx >> 8;
    int n = idx & 255;
    Wt[(size_t)n * K + k] = (__bf16)W[idx];
}

// ---------------------------------------------------------------------------
// WMMA GEMM. grid.x = M/(16*MT) row strips, 128 threads = 4 waves.
// Wave w computes N-tiles 4w..4w+3 (covers N=256) for all MT M-tiles.
// K = K1+K2 (multiple of 32). A1 rows optionally gathered through gidx.
// Wt: bf16 [256,K] transposed weights -> B fragment = 2 contiguous 16B loads.
// ---------------------------------------------------------------------------
template <int MT>
__global__ __launch_bounds__(128)
void gemm_wmma(const float* __restrict__ A1, const int* __restrict__ gidx, int K1,
               const float* __restrict__ A2, int K2,
               const __bf16* __restrict__ Wt, const float* __restrict__ bias,
               float* __restrict__ C, int doRelu)
{
    __shared__ __attribute__((aligned(16))) __bf16 As[2][MT * 16][40]; // padded rows

    const int tid  = threadIdx.x;
    const int lane = tid & 31;
    const int wave = tid >> 5;
    const int m0   = blockIdx.x * (MT * 16);
    const int K    = K1 + K2;

    v8f acc[MT][4];
#pragma unroll
    for (int mt = 0; mt < MT; ++mt)
#pragma unroll
        for (int t = 0; t < 4; ++t)
#pragma unroll
            for (int i = 0; i < 8; ++i) acc[mt][t][i] = 0.0f;

    const int mA  = lane & 15;            // A row within tile
    const int kh  = (lane >> 4) << 3;     // 0 or 8 (K half select)
    const int nlo = lane & 15;            // B/C column within tile

    // staging coords: MT float4's per thread (MT*512 elems / 128 threads)
    const int sr = tid >> 3;              // 0..15
    const int sc = (tid & 7) << 2;        // 0,4,...,28

    auto stage = [&](int kb, int buf) {
#pragma unroll
        for (int q = 0; q < MT; ++q) {
            int r = sr + 16 * q;
            int grow = m0 + r;
            int k = kb + sc;              // K1 multiple of 32 -> no straddle
            float4 v;
            if (k < K1) {
                int ar = gidx ? gidx[grow] : grow;
                v = *(const float4*)(A1 + (size_t)ar * K1 + k);
            } else {
                v = *(const float4*)(A2 + (size_t)grow * K2 + (k - K1));
            }
            v4bf pk;
            pk[0] = (__bf16)v.x; pk[1] = (__bf16)v.y;
            pk[2] = (__bf16)v.z; pk[3] = (__bf16)v.w;
            *(v4bf*)&As[buf][r][sc] = pk;
        }
    };

    stage(0, 0);
    __syncthreads();

    for (int kb = 0; kb < K; kb += 32) {
        const int buf = (kb >> 5) & 1;

        // batch-load all B fragments for this K-step (8 independent b128 loads)
        v16bf bfrag[4];
#pragma unroll
        for (int t = 0; t < 4; ++t) {
            int n = ((wave << 2) + t) * 16 + nlo;
            const __bf16* wr = Wt + (size_t)n * K + kb;
            v8bf blo = *(const v8bf*)(wr + kh);
            v8bf bhi = *(const v8bf*)(wr + 16 + kh);
            bfrag[t] = __builtin_shufflevector(blo, bhi,
                0, 1, 2, 3, 4, 5, 6, 7, 8, 9, 10, 11, 12, 13, 14, 15);
        }

        // A fragments: two 16B LDS loads per M-tile
        v16bf afrag[MT];
#pragma unroll
        for (int mt = 0; mt < MT; ++mt) {
            v8bf alo = *(const v8bf*)&As[buf][mt * 16 + mA][kh];
            v8bf ahi = *(const v8bf*)&As[buf][mt * 16 + mA][16 + kh];
            afrag[mt] = __builtin_shufflevector(alo, ahi,
                0, 1, 2, 3, 4, 5, 6, 7, 8, 9, 10, 11, 12, 13, 14, 15);
        }

        // WMMA chain: MT*4 matrix ops, B fragments reused across M-tiles
#pragma unroll
        for (int t = 0; t < 4; ++t)
#pragma unroll
            for (int mt = 0; mt < MT; ++mt)
                acc[mt][t] = __builtin_amdgcn_wmma_f32_16x16x32_bf16(
                    false, afrag[mt], false, bfrag[t], (short)0, acc[mt][t], false, false);

        if (kb + 32 < K) stage(kb + 32, buf ^ 1);   // overlap next-stage with WMMAs
        __syncthreads();
    }

    // epilogue: C layout lane n = lane&15, rows (lane>=16?8:0)+r
    const int rhi = (lane >> 4) << 3;
#pragma unroll
    for (int mt = 0; mt < MT; ++mt)
#pragma unroll
        for (int t = 0; t < 4; ++t) {
            int col = ((wave << 2) + t) * 16 + nlo;
            float bv = bias ? bias[col] : 0.0f;
#pragma unroll
            for (int r = 0; r < 8; ++r) {
                int row = m0 + mt * 16 + rhi + r;
                float v = acc[mt][t][r] + bv;
                if (doRelu) v = fmaxf(v, 0.0f);
                C[(size_t)row * HID + col] = v;
            }
        }
}

// ---------------------------------------------------------------------------
__global__ void zero_kernel(float* __restrict__ p, int n)
{
    int i = blockIdx.x * blockDim.x + threadIdx.x;
    if (i < n) p[i] = 0.0f;
}

__global__ void scatter_kernel(const float* __restrict__ m, const int* __restrict__ dst,
                               float* __restrict__ agg)
{
    int idx = blockIdx.x * blockDim.x + threadIdx.x;   // E*256 threads
    int e = idx >> 8;
    int c = idx & 255;
    atomicAdd(&agg[(size_t)dst[e] * HID + c], m[idx]);
}

// LayerNorm + (optional residual) + relu, one block per row.
__global__ __launch_bounds__(256)
void ln_relu_kernel(const float* __restrict__ xin, const float* __restrict__ res,
                    const float* __restrict__ gamma, const float* __restrict__ beta,
                    float* __restrict__ xout, int hasRes)
{
    __shared__ float red[256];
    int r = blockIdx.x, c = threadIdx.x;
    float v = xin[(size_t)r * HID + c];
    red[c] = v;
    __syncthreads();
    for (int s = 128; s > 0; s >>= 1) { if (c < s) red[c] += red[c + s]; __syncthreads(); }
    float mu = red[0] * (1.0f / 256.0f);
    __syncthreads();
    float d = v - mu;
    red[c] = d * d;
    __syncthreads();
    for (int s = 128; s > 0; s >>= 1) { if (c < s) red[c] += red[c + s]; __syncthreads(); }
    float var = red[0] * (1.0f / 256.0f);
    float y = d * rsqrtf(var + 1e-5f) * gamma[c] + beta[c];
    if (hasRes) y += res[(size_t)r * HID + c];
    xout[(size_t)r * HID + c] = fmaxf(y, 0.0f);
}

// g = mean_rows(x) @ Wg + bg   (single block of 256 threads)
__global__ __launch_bounds__(256)
void glob_kernel(const float* __restrict__ x, const float* __restrict__ Wg,
                 const float* __restrict__ bg, float* __restrict__ g)
{
    __shared__ float xm[256];
    int c = threadIdx.x;
    float s = 0.0f;
    for (int r = 0; r < NNODE; ++r) s += x[(size_t)r * HID + c];
    xm[c] = s * (1.0f / (float)NNODE);
    __syncthreads();
    float acc = bg[c];
    for (int k = 0; k < HID; ++k) acc += xm[k] * Wg[(size_t)k * HID + c];
    g[c] = acc;
}

// outv = g @ W[rowoff:rowoff+256] + bias   (broadcast-context folded to bias)
__global__ __launch_bounds__(256)
void gvec_kernel(const float* __restrict__ g, const float* __restrict__ W, int rowoff,
                 const float* __restrict__ bias, float* __restrict__ outv)
{
    __shared__ float gs[256];
    int c = threadIdx.x;
    gs[c] = g[c];
    __syncthreads();
    float acc = bias[c];
    for (int k = 0; k < HID; ++k) acc += gs[k] * W[(size_t)(rowoff + k) * HID + c];
    outv[c] = acc;
}

__global__ void mv_head(const float* __restrict__ z, const float* __restrict__ W2,
                        const float* __restrict__ b2, float* __restrict__ out_mv,
                        float* __restrict__ out_soft)
{
    int i = blockIdx.x * blockDim.x + threadIdx.x;
    if (i >= NNODE) return;
    float a0 = b2[0], a1 = b2[1], a2 = b2[2], a3 = b2[3];
    const float* zr = z + (size_t)i * HID;
    for (int h = 0; h < HID; ++h) {
        float zv = zr[h];
        a0 += zv * W2[h * 4 + 0];
        a1 += zv * W2[h * 4 + 1];
        a2 += zv * W2[h * 4 + 2];
        a3 += zv * W2[h * 4 + 3];
    }
    out_mv[i * 4 + 0] = a0; out_mv[i * 4 + 1] = a1;
    out_mv[i * 4 + 2] = a2; out_mv[i * 4 + 3] = a3;
    float mx = fmaxf(fmaxf(a0, a1), fmaxf(a2, a3));
    float e0 = expf(a0 - mx), e1 = expf(a1 - mx), e2 = expf(a2 - mx), e3 = expf(a3 - mx);
    float den = 1.0f / (e0 + e1 + e2 + e3);
    out_soft[i * 4 + 0] = e0 * den; out_soft[i * 4 + 1] = e1 * den;
    out_soft[i * 4 + 2] = e2 * den; out_soft[i * 4 + 3] = e3 * den;
}

__global__ void sv_head(const float* __restrict__ z, const float* __restrict__ w2,
                        const float* __restrict__ b2, float* __restrict__ out_sv)
{
    int i = blockIdx.x * blockDim.x + threadIdx.x;
    if (i >= NNODE) return;
    float s = b2[0];
    const float* zr = z + (size_t)i * HID;
    for (int h = 0; h < HID; ++h) s += zr[h] * w2[h];
    out_sv[i] = fminf(fmaxf(s, -10.0f), 10.0f);
}

// ts[i, j!=i] = clip( relu(zi[i]+zj[j]) . w2 + b2 )  — relu inside the dot, VALU
__global__ __launch_bounds__(256)
void pair_kernel(const float* __restrict__ zi, const float* __restrict__ zj,
                 const float* __restrict__ w2, const float* __restrict__ b2,
                 float* __restrict__ ts)
{
    __shared__ float zis[256], ws[256];
    int i = blockIdx.x;
    int t = threadIdx.x;
    zis[t] = zi[(size_t)i * HID + t];
    ws[t]  = w2[t];
    __syncthreads();
    float bb = b2[0];
    for (int j = t; j < NNODE; j += 256) {
        const float* zr = zj + (size_t)j * HID;
        float s = bb;
        for (int h = 0; h < HID; ++h)
            s += fmaxf(zis[h] + zr[h], 0.0f) * ws[h];
        s = fminf(fmaxf(s, -10.0f), 10.0f);
        if (j != i) {
            int col = (j < i) ? j : j - 1;
            ts[(size_t)i * (NNODE - 1) + col] = s;
        }
    }
}

// ---------------------------------------------------------------------------
// d_in layout (JAX pytree order, dict keys sorted):
//  0 node_features  1 edge_indices i32  2 edge_features
//  3 glob.b  4 glob.w
//  5 + 12*l : edge.b, edge.w, m1.b, m1.w, m2.b, m2.w, node.b, node.w, u1.b, u1.w, u2.b, u2.w
//  77 + 2*l : ln.b, ln.g
//  89 move1.b 90 move1.w 91 move2.b 92 move2.w 93 sh1.b 94 sh1.w
//  95 sh2.b 96 sh2.w 97 tg1.b 98 tg1.w 99 tg2.b 100 tg2.w
// ---------------------------------------------------------------------------
static inline void convert_wt(const float* W, __bf16* Wt, int K, hipStream_t s)
{
    int n = K * HID;
    convert_wt_kernel<<<(n + 255) / 256, 256, 0, s>>>(W, Wt, K);
}

extern "C" void kernel_launch(void* const* d_in, const int* in_sizes, int n_in,
                              void* d_out, int out_size, void* d_ws, size_t ws_size,
                              hipStream_t stream)
{
    (void)in_sizes; (void)n_in; (void)out_size; (void)ws_size;

    const float* nodef = (const float*)d_in[0];
    const int*   eidx  = (const int*)d_in[1];
    const int*   srcp  = eidx;
    const int*   dstp  = eidx + NEDGE;
    const float* ef    = (const float*)d_in[2];
    const float* glob_b = (const float*)d_in[3];
    const float* glob_w = (const float*)d_in[4];
    const float* move1_b = (const float*)d_in[89];
    const float* move1_w = (const float*)d_in[90];
    const float* move2_b = (const float*)d_in[91];
    const float* move2_w = (const float*)d_in[92];
    const float* sh1_b   = (const float*)d_in[93];
    const float* sh1_w   = (const float*)d_in[94];
    const float* sh2_b   = (const float*)d_in[95];
    const float* sh2_w   = (const float*)d_in[96];
    const float* tg1_b   = (const float*)d_in[97];
    const float* tg1_w   = (const float*)d_in[98];
    const float* tg2_b   = (const float*)d_in[99];
    const float* tg2_w   = (const float*)d_in[100];

    // ---- workspace carve: fp32 activations, then bf16 weight scratch ----
    float* p = (float*)d_ws;
    float* buf_e = p; p += (size_t)NEDGE * HID;   // e -> t1 -> m (in-place: 1:1 row strips)
    float* x_a   = p; p += (size_t)NNODE * HID;
    float* x_b   = p; p += (size_t)NNODE * HID;
    float* hbuf  = p; p += (size_t)NNODE * HID;
    float* agg   = p; p += (size_t)NNODE * HID;
    float* t2    = p; p += (size_t)NNODE * HID;
    float* xnew  = p; p += (size_t)NNODE * HID;
    float* hd1   = p; p += (size_t)NNODE * HID;
    float* hd2   = p; p += (size_t)NNODE * HID;
    float* gbuf  = p; p += HID;
    float* gvec  = p; p += HID * 3;               // keep cursor 16B aligned

    __bf16* q = (__bf16*)p;                       // bf16 weight scratch (reused per layer)
    __bf16* wt_node = q; q += 256 * HID;
    __bf16* wt_edge = q; q += 32 * HID;
    __bf16* wt_m1   = q; q += 512 * HID;
    __bf16* wt_m2   = q; q += 256 * HID;
    __bf16* wt_u1   = q; q += 512 * HID;
    __bf16* wt_u2   = q; q += 256 * HID;
    __bf16* wt_hA   = q; q += 256 * HID;          // head slice A
    __bf16* wt_hB   = q; q += 256 * HID;          // head slice B

    float* out      = (float*)d_out;
    float* out_mv   = out;                                  // [512,4]
    float* out_soft = out + NNODE * 4;                      // [512,4]
    float* out_sv   = out + NNODE * 8;                      // [512]
    float* out_ts   = out + NNODE * 8 + NNODE;              // [512*511]
    float* out_x    = out_ts + (size_t)NNODE * (NNODE - 1); // [512,256]

    const float* x_cur = nodef;
    int Din = 64;
    for (int l = 0; l < 6; ++l) {
        const float* e_b  = (const float*)d_in[5 + l * 12 + 0];
        const float* e_w  = (const float*)d_in[5 + l * 12 + 1];
        const float* m1_b = (const float*)d_in[5 + l * 12 + 2];
        const float* m1_w = (const float*)d_in[5 + l * 12 + 3];
        const float* m2_b = (const float*)d_in[5 + l * 12 + 4];
        const float* m2_w = (const float*)d_in[5 + l * 12 + 5];
        const float* n_b  = (const float*)d_in[5 + l * 12 + 6];
        const float* n_w  = (const float*)d_in[5 + l * 12 + 7];
        const float* u1_b = (const float*)d_in[5 + l * 12 + 8];
        const float* u1_w = (const float*)d_in[5 + l * 12 + 9];
        const float* u2_b = (const float*)d_in[5 + l * 12 + 10];
        const float* u2_w = (const float*)d_in[5 + l * 12 + 11];
        const float* ln_b = (const float*)d_in[77 + l * 2 + 0];
        const float* ln_g = (const float*)d_in[77 + l * 2 + 1];
        float* x_next = (l & 1) ? x_b : x_a;

        // weight conversion (transpose fp32 -> bf16 [N,K])
        convert_wt(n_w,  wt_node, Din,       stream);
        convert_wt(e_w,  wt_edge, 32,        stream);
        convert_wt(m1_w, wt_m1,   512,       stream);
        convert_wt(m2_w, wt_m2,   256,       stream);
        convert_wt(u1_w, wt_u1,   Din + 256, stream);
        convert_wt(u2_w, wt_u2,   256,       stream);

        // h = x @ Wn + bn
        gemm_wmma<1><<<NNODE / 16, 128, 0, stream>>>(x_cur, nullptr, Din, nullptr, 0, wt_node, n_b, hbuf, 0);
        // e = ef @ We + be
        gemm_wmma<4><<<NEDGE / 64, 128, 0, stream>>>(ef, nullptr, 32, nullptr, 0, wt_edge, e_b, buf_e, 0);
        // t1 = relu( [h[src], e] @ Wm1 + bm1 )   (in-place over buf_e)
        gemm_wmma<4><<<NEDGE / 64, 128, 0, stream>>>(hbuf, srcp, HID, buf_e, HID, wt_m1, m1_b, buf_e, 1);
        // m = t1 @ Wm2 + bm2                     (in-place)
        gemm_wmma<4><<<NEDGE / 64, 128, 0, stream>>>(buf_e, nullptr, HID, nullptr, 0, wt_m2, m2_b, buf_e, 0);
        // agg = scatter_add(m by dst)
        zero_kernel<<<(NNODE * HID) / 256, 256, 0, stream>>>(agg, NNODE * HID);
        scatter_kernel<<<NEDGE, 256, 0, stream>>>(buf_e, dstp, agg);
        // t2 = relu( [res, agg] @ Wu1 + bu1 )
        gemm_wmma<1><<<NNODE / 16, 128, 0, stream>>>(x_cur, nullptr, Din, agg, HID, wt_u1, u1_b, t2, 1);
        // xnew = t2 @ Wu2 + bu2
        gemm_wmma<1><<<NNODE / 16, 128, 0, stream>>>(t2, nullptr, HID, nullptr, 0, wt_u2, u2_b, xnew, 0);
        // x = relu( LN(xnew) [+ res] )
        ln_relu_kernel<<<NNODE, 256, 0, stream>>>(xnew, x_cur, ln_g, ln_b, x_next, (Din == HID) ? 1 : 0);

        x_cur = x_next;
        Din = HID;
    }

    // ---- heads ----
    glob_kernel<<<1, 256, 0, stream>>>(x_cur, glob_w, glob_b, gbuf);

    // move head: bias' = g @ Wmv1[256:512] + bmv1 ; zmv = relu(x @ Wmv1[:256] + bias')
    convert_wt(move1_w, wt_hA, 256, stream);
    gvec_kernel<<<1, 256, 0, stream>>>(gbuf, move1_w, 256, move1_b, gvec);
    gemm_wmma<1><<<NNODE / 16, 128, 0, stream>>>(x_cur, nullptr, HID, nullptr, 0, wt_hA, gvec, hd1, 1);
    mv_head<<<2, 256, 0, stream>>>(hd1, move2_w, move2_b, out_mv, out_soft);

    // shoot head
    convert_wt(sh1_w, wt_hA, 256, stream);
    gvec_kernel<<<1, 256, 0, stream>>>(gbuf, sh1_w, 256, sh1_b, gvec);
    gemm_wmma<1><<<NNODE / 16, 128, 0, stream>>>(x_cur, nullptr, HID, nullptr, 0, wt_hA, gvec, hd1, 1);
    sv_head<<<2, 256, 0, stream>>>(hd1, sh2_w, sh2_b, out_sv);

    // target head: zi = x@W1[:256] + (g@W1[512:] + b1) ; zj = x@W1[256:512]
    convert_wt(tg1_w,             wt_hA, 256, stream);
    convert_wt(tg1_w + 256 * HID, wt_hB, 256, stream);
    gvec_kernel<<<1, 256, 0, stream>>>(gbuf, tg1_w, 512, tg1_b, gvec);
    gemm_wmma<1><<<NNODE / 16, 128, 0, stream>>>(x_cur, nullptr, HID, nullptr, 0, wt_hA, gvec, hd1, 0);
    gemm_wmma<1><<<NNODE / 16, 128, 0, stream>>>(x_cur, nullptr, HID, nullptr, 0, wt_hB, nullptr, hd2, 0);
    pair_kernel<<<NNODE, 256, 0, stream>>>(hd1, hd2, tg2_w, tg2_b, out_ts);

    // final node embeddings
    hipMemcpyAsync(out_x, x_cur, (size_t)NNODE * HID * sizeof(float),
                   hipMemcpyDeviceToDevice, stream);
}
